// MultiOrderWeightedConv1D2_44349832299093
// MI455X (gfx1250) — matrix operand order repaired and checked by reading, not compiled, not run on hardware
//
#include <hip/hip_runtime.h>
#include <hip/hip_bf16.h>
#include <stdint.h>

typedef _Float16 hf;
typedef __attribute__((ext_vector_type(16))) _Float16 v16h;
typedef __attribute__((ext_vector_type(8)))  _Float16 v8h;
typedef __attribute__((ext_vector_type(4)))  _Float16 v4h;
typedef __attribute__((ext_vector_type(8)))  float    v8f;
typedef __attribute__((ext_vector_type(4)))  unsigned int v4u;

#define B_     8
#define CIN    128
#define COUT   128
#define NPTS   16384
#define TROT   2
#define OUTC   256
#define KW     9
#define KDIM   (CIN*KW)      // 1152 = 36 * 32
#define PADW   4
#define SIGMA_ 0.05f
#define RES_   128
#define NT     128
#define HALO   8
#define LDA    40            // padded LDS row stride (halfs): 80B -> conflict-free b128 reads

static __device__ __forceinline__ v8f wmma_f16(v16h a, v16h b, v8f c) {
  // v_wmma_f32_16x16x32_f16  (8-arg form: neg_a, A, neg_b, B, c_mod, C, reuse_a, reuse_b)
  return __builtin_amdgcn_wmma_f32_16x16x32_f16(false, a, false, b, (short)0, c, false, false);
}

// two aligned 16B LDS loads -> one 16-half fragment (ds_load_b128 x2)
static __device__ __forceinline__ v16h ld_frag(const hf* p0, const hf* p1) {
  v8h lo = *(const v8h*)p0;
  v8h hi = *(const v8h*)p1;
  return __builtin_shufflevector(lo, hi, 0, 1, 2, 3, 4, 5, 6, 7,
                                 8, 9, 10, 11, 12, 13, 14, 15);
}

// ---------------------------------------------------------------- utilities
__global__ void cvt_f32_f16(const float* __restrict__ s, hf* __restrict__ d, int n) {
  int i = blockIdx.x * 256 + threadIdx.x;
  if (i < n) d[i] = (hf)s[i];
}

// ------------------------------------------------- rotated-coordinate min/max
__global__ __launch_bounds__(256) void minmax_kernel(const float* __restrict__ coords,
                                                     const float* __restrict__ rot,
                                                     float* __restrict__ mm) {
  __shared__ float red[256 * 6];
  int b = blockIdx.x, tid = threadIdx.x;
  float r[9];
#pragma unroll
  for (int i = 0; i < 9; ++i) r[i] = rot[i];
  float mn[3] = {3.4e38f, 3.4e38f, 3.4e38f};
  float mx[3] = {-3.4e38f, -3.4e38f, -3.4e38f};
  const float* c0p = coords + ((size_t)b * 3 + 0) * NPTS;
  const float* c1p = coords + ((size_t)b * 3 + 1) * NPTS;
  const float* c2p = coords + ((size_t)b * 3 + 2) * NPTS;
  for (int n = tid; n < NPTS; n += 256) {
    float c0 = c0p[n], c1 = c1p[n], c2 = c2p[n];
#pragma unroll
    for (int d = 0; d < 3; ++d) {
      float rp = c0 * r[d] + c1 * r[3 + d] + c2 * r[6 + d];
      mn[d] = fminf(mn[d], rp);
      mx[d] = fmaxf(mx[d], rp);
    }
  }
#pragma unroll
  for (int d = 0; d < 3; ++d) { red[tid * 6 + d] = mn[d]; red[tid * 6 + 3 + d] = mx[d]; }
  __syncthreads();
  for (int s = 128; s > 0; s >>= 1) {
    if (tid < s) {
#pragma unroll
      for (int d = 0; d < 3; ++d) {
        red[tid * 6 + d]     = fminf(red[tid * 6 + d],     red[(tid + s) * 6 + d]);
        red[tid * 6 + 3 + d] = fmaxf(red[tid * 6 + 3 + d], red[(tid + s) * 6 + 3 + d]);
      }
    }
    __syncthreads();
  }
  if (tid < 6) mm[b * 6 + tid] = red[tid];
}

// ---------------------------------------------- hilbert key (packed with idx)
__global__ void keys_kernel(const float* __restrict__ coords, const float* __restrict__ rot,
                            const float* __restrict__ mm, const int* __restrict__ dist,
                            unsigned long long* __restrict__ keys) {
  int i = blockIdx.x * 256 + threadIdx.x;
  if (i >= B_ * NPTS) return;
  int b = i >> 14;
  int n = i & (NPTS - 1);
  float c0 = coords[((size_t)b * 3 + 0) * NPTS + n];
  float c1 = coords[((size_t)b * 3 + 1) * NPTS + n];
  float c2 = coords[((size_t)b * 3 + 2) * NPTS + n];
  int cell = 0;
#pragma unroll
  for (int d = 0; d < 3; ++d) {
    float rp  = c0 * rot[d] + c1 * rot[3 + d] + c2 * rot[6 + d];
    float mn  = mm[b * 6 + d], mx = mm[b * 6 + 3 + d];
    float nrm = (rp - mn) / ((mx - mn) + 1e-23f);
    int q = (int)floorf(nrm * (float)(RES_ - 1));
    q = q < 0 ? 0 : (q > RES_ - 1 ? RES_ - 1 : q);
    cell = cell * RES_ + q;
  }
  unsigned hd = (unsigned)dist[cell];
  keys[i] = ((unsigned long long)hd << 32) | (unsigned)n;
}

// ------------------------------------------------------- global bitonic sort
__global__ void bitonic_step(unsigned long long* __restrict__ keys, int k, int j) {
  int i = blockIdx.x * 256 + threadIdx.x;
  if (i >= B_ * NPTS) return;
  int il = i & (NPTS - 1);
  int l = il ^ j;
  if (l > il) {
    unsigned long long* base = keys + (i - il);
    unsigned long long a = base[il], c = base[l];
    bool asc = (il & k) == 0;
    if ((a > c) == asc) { base[il] = c; base[l] = a; }
  }
}

__global__ void idx_kernel(const unsigned long long* __restrict__ keys, int* __restrict__ idx) {
  int i = blockIdx.x * 256 + threadIdx.x;
  if (i >= B_ * NPTS) return;
  idx[i] = (int)(unsigned)(keys[i] & 0xffffffffULL);
}

// --------------------------------------------------------------- gathers
__global__ void gather_feats(const float* __restrict__ x, const int* __restrict__ idx,
                             hf* __restrict__ feats) {
  size_t i = (size_t)blockIdx.x * 256 + threadIdx.x;
  if (i >= (size_t)B_ * CIN * NPTS) return;
  int pos = (int)(i & (NPTS - 1));
  size_t bc = i >> 14;
  int b = (int)(bc >> 7);
  int n = idx[(size_t)b * NPTS + pos];
  feats[i] = (hf)x[(bc << 14) + n];
}

__global__ void gather_xyz(const float* __restrict__ coords, const int* __restrict__ idx,
                           float* __restrict__ xyz) {
  int i = blockIdx.x * 256 + threadIdx.x;
  if (i >= B_ * 3 * NPTS) return;
  int pos = i & (NPTS - 1);
  int bd = i >> 14;
  int b = bd / 3;
  int n = idx[(size_t)b * NPTS + pos];
  xyz[i] = coords[((size_t)bd << 14) + n];
}

// ---------------------------------- masked-window conv as implicit WMMA GEMM
// D[o,n] = sum_{i,k} W[o, i*9+k] * feats[i, n+k-4] * mask[k, n]
__global__ __launch_bounds__(256) void conv_wmma(const hf* __restrict__ feats,
                                                 const float* __restrict__ xyz,
                                                 const hf* __restrict__ w,
                                                 float* __restrict__ rbuf) {
  __shared__ hf    sF[CIN][NT + HALO];   // 34816 B
  __shared__ float sM[KW][NT];           //  4608 B
  __shared__ float sX[3][NT + HALO];     //  1632 B
  __shared__ hf    sA[COUT * LDA];       // 10240 B (weights k-chunk, [m][kk], padded)
  __shared__ hf    sB[NT * LDA];         // 10240 B (masked windows, [n][kk], padded)

  int b  = blockIdx.y;
  int n0 = blockIdx.x * NT;
  int tid = threadIdx.x;

  for (int q = tid; q < 3 * (NT + HALO); q += 256) {
    int d = q / (NT + HALO), p = q % (NT + HALO);
    int g = n0 + p - PADW;
    sX[d][p] = (g >= 0 && g < NPTS) ? xyz[((size_t)b * 3 + d) * NPTS + g] : 0.f;
  }
  if (n0 >= PADW && n0 + NT + PADW <= NPTS) {
    // interior tile: vectorized 8B copies (34 x v4h per row)
    for (int q = tid; q < CIN * 34; q += 256) {
      int c = q / 34, p = q % 34;
      *(v4h*)&sF[c][p * 4] =
          *(const v4h*)(feats + ((size_t)b * CIN + c) * NPTS + n0 - PADW + p * 4);
    }
  } else {
    for (int q = tid; q < CIN * (NT + HALO); q += 256) {
      int c = q / (NT + HALO), p = q % (NT + HALO);
      int g = n0 + p - PADW;
      sF[c][p] = (g >= 0 && g < NPTS) ? feats[((size_t)b * CIN + c) * NPTS + g] : (hf)0.f;
    }
  }
  __syncthreads();
  for (int q = tid; q < KW * NT; q += 256) {
    int k = q >> 7, n = q & (NT - 1);
    float dx = sX[0][n + k] - sX[0][n + PADW];
    float dy = sX[1][n + k] - sX[1][n + PADW];
    float dz = sX[2][n + k] - sX[2][n + PADW];
    float dist = sqrtf(dx * dx + dy * dy + dz * dz + 1e-20f);
    float m = 1.f - dist * (1.f / SIGMA_);
    sM[k][n] = m > 0.f ? m : 0.f;
  }
  __syncthreads();

  int lane = tid & 31, wave = tid >> 5;
  int m0 = wave * 16;
  v8f acc[8];
  v8f zed = {0.f, 0.f, 0.f, 0.f, 0.f, 0.f, 0.f, 0.f};
#pragma unroll
  for (int i = 0; i < 8; ++i) acc[i] = zed;

  for (int kc = 0; kc < KDIM / 32; ++kc) {
    int k0 = kc * 32;
    // prefetch next weight k-chunk into cache (global_prefetch_b8)
    if (kc + 1 < KDIM / 32 && tid < COUT)
      __builtin_prefetch(w + (size_t)tid * KDIM + k0 + 32, 0, 1);
    // stage weights chunk: COUT rows x 32 halfs as b128 copies
    for (int q = tid; q < COUT * 4; q += 256) {
      int m = q >> 2, xq = q & 3;
      *((v4u*)(sA + m * LDA) + xq) = ((const v4u*)(w + (size_t)m * KDIM + k0))[xq];
    }
    // stage masked-window chunk: NT cols x 32 K-halfs, built once, shared by 8 waves
    for (int q = tid; q < NT * 32; q += 256) {
      int n = q >> 5, kk = q & 31;
      int kap = k0 + kk;
      int ci = kap / KW;
      int kw = kap - ci * KW;
      sB[n * LDA + kk] = (hf)((float)sF[ci][n + kw] * sM[kw][n]);
    }
    __syncthreads();
    // A frag: row = m0 + lane&15; lanes 0-15: K 0-7 & 16-23, lanes 16-31: K 8-15 & 24-31
    const hf* pa = &sA[(m0 + (lane & 15)) * LDA + ((lane & 16) ? 8 : 0)];
    v16h af = ld_frag(pa, pa + 16);
#pragma unroll
    for (int ns = 0; ns < 8; ++ns) {
      // B frag: col = lane&15; lanes 0-15: K 0-7 & 16-23, lanes 16-31: K 8-15 & 24-31
      const hf* pb = &sB[(ns * 16 + (lane & 15)) * LDA + ((lane & 16) ? 8 : 0)];
      v16h bf = ld_frag(pb, pb + 16);
      acc[ns] = wmma_f16(af, bf, acc[ns]);
    }
    __syncthreads();
  }
  int moff = (lane & 16) ? 8 : 0, ncol = lane & 15;
#pragma unroll
  for (int ns = 0; ns < 8; ++ns)
#pragma unroll
    for (int r = 0; r < 8; ++r)
      rbuf[((size_t)b * COUT + m0 + r + moff) * NPTS + n0 + ns * 16 + ncol] = acc[ns][r];
}

// --------------------------------------------------------------- batch norm
__global__ __launch_bounds__(256) void bn_stats(const float* __restrict__ rbuf,
                                                float* __restrict__ stats) {
  __shared__ float rs[256], rs2[256];
  int o = blockIdx.x, tid = threadIdx.x;
  float s = 0.f, s2 = 0.f;
  for (int i = tid; i < B_ * NPTS; i += 256) {
    int b = i >> 14, n = i & (NPTS - 1);
    float v = rbuf[((size_t)b * COUT + o) * NPTS + n];
    s += v; s2 += v * v;
  }
  rs[tid] = s; rs2[tid] = s2;
  __syncthreads();
  for (int st = 128; st > 0; st >>= 1) {
    if (tid < st) { rs[tid] += rs[tid + st]; rs2[tid] += rs2[tid + st]; }
    __syncthreads();
  }
  if (tid == 0) {
    float cnt = (float)(B_ * NPTS);
    float mean = rs[0] / cnt;
    float var = rs2[0] / cnt - mean * mean;
    stats[o * 2] = mean;
    stats[o * 2 + 1] = rsqrtf(var + 1e-5f);
  }
}

__global__ void bn_scatter(const float* __restrict__ rbuf, const float* __restrict__ stats,
                           const float* __restrict__ gamma, const float* __restrict__ beta,
                           const int* __restrict__ idx, hf* __restrict__ ofeat, int t) {
  size_t i = (size_t)blockIdx.x * 256 + threadIdx.x;
  if (i >= (size_t)B_ * COUT * NPTS) return;
  int pos = (int)(i & (NPTS - 1));
  size_t bo = i >> 14;
  int o = (int)(bo & (COUT - 1));
  int b = (int)(bo >> 7);
  float v = (rbuf[i] - stats[o * 2]) * stats[o * 2 + 1] * gamma[o] + beta[o];
  int n = idx[(size_t)b * NPTS + pos];
  ofeat[((size_t)b * OUTC + t * COUT + o) * NPTS + n] = (hf)v;
}

// ------------------------------------------------- pointwise GEMM (WMMA)
__global__ __launch_bounds__(256) void pw_wmma(const hf* __restrict__ ofeat,
                                               const hf* __restrict__ pw,
                                               const float* __restrict__ bias,
                                               float* __restrict__ out) {
  __shared__ hf sA[OUTC * LDA];  // 20480 B
  __shared__ hf sB[64 * LDA];    //  5120 B
  int b = blockIdx.y;
  int n0 = blockIdx.x * 64;
  int tid = threadIdx.x, lane = tid & 31, wave = tid >> 5;
  v8f acc[2][4];
  v8f zed = {0.f, 0.f, 0.f, 0.f, 0.f, 0.f, 0.f, 0.f};
#pragma unroll
  for (int i = 0; i < 2; ++i)
#pragma unroll
    for (int j = 0; j < 4; ++j) acc[i][j] = zed;

  for (int kc = 0; kc < OUTC / 32; ++kc) {
    int k0 = kc * 32;
    // stage weights: OUTC rows x 32 halfs as b128 copies
    for (int q = tid; q < OUTC * 4; q += 256)
      *((v4u*)(sA + (q >> 2) * LDA) + (q & 3)) =
          ((const v4u*)(pw + (size_t)(q >> 2) * OUTC + k0))[q & 3];
    // stage activations: [n][kk], 8B global loads
    for (int q = tid; q < 32 * 16; q += 256) {
      int kk = q >> 4, nc = q & 15;
      v4h v = *(const v4h*)(ofeat + ((size_t)b * OUTC + k0 + kk) * NPTS + n0 + nc * 4);
#pragma unroll
      for (int e = 0; e < 4; ++e) sB[(nc * 4 + e) * LDA + kk] = v[e];
    }
    __syncthreads();
    int kb = (lane & 16) ? 8 : 0;
    const hf* pa0 = &sA[(wave * 32 + (lane & 15)) * LDA + kb];
    const hf* pa1 = &sA[(wave * 32 + 16 + (lane & 15)) * LDA + kb];
    v16h af0 = ld_frag(pa0, pa0 + 16);
    v16h af1 = ld_frag(pa1, pa1 + 16);
#pragma unroll
    for (int ns = 0; ns < 4; ++ns) {
      const hf* pb = &sB[(ns * 16 + (lane & 15)) * LDA + ((lane & 16) ? 8 : 0)];
      v16h bf = ld_frag(pb, pb + 16);
      acc[0][ns] = wmma_f16(af0, bf, acc[0][ns]);
      acc[1][ns] = wmma_f16(af1, bf, acc[1][ns]);
    }
    __syncthreads();
  }
  int moff = (lane & 16) ? 8 : 0, ncol = lane & 15;
#pragma unroll
  for (int ms = 0; ms < 2; ++ms)
#pragma unroll
    for (int ns = 0; ns < 4; ++ns)
#pragma unroll
      for (int r = 0; r < 8; ++r) {
        int m = wave * 32 + ms * 16 + r + moff;
        out[((size_t)b * OUTC + m) * NPTS + n0 + ns * 16 + ncol] = acc[ms][ns][r] + bias[m];
      }
}

// ----------------------------------------------------------------- host side
extern "C" void kernel_launch(void* const* d_in, const int* in_sizes, int n_in,
                              void* d_out, int out_size, void* d_ws, size_t ws_size,
                              hipStream_t stream) {
  (void)in_sizes; (void)n_in; (void)out_size; (void)ws_size;
  const float* x      = (const float*)d_in[0];
  const float* coords = (const float*)d_in[1];
  const float* rots   = (const float*)d_in[2];
  const int*   dist   = (const int*)d_in[3];
  const float* convw  = (const float*)d_in[4];
  const float* gamma  = (const float*)d_in[5];
  const float* beta   = (const float*)d_in[6];
  const float* pww    = (const float*)d_in[7];
  const float* pwb    = (const float*)d_in[8];
  float* out = (float*)d_out;

  char* ws = (char*)d_ws;
  size_t off = 0;
  auto take = [&](size_t bytes) -> char* {
    char* p = ws + off;
    off += (bytes + 255) & ~(size_t)255;
    return p;
  };
  hf*    wf16   = (hf*)take((size_t)COUT * KDIM * sizeof(hf));
  hf*    pwf16  = (hf*)take((size_t)OUTC * OUTC * sizeof(hf));
  float* mm     = (float*)take((size_t)B_ * 6 * sizeof(float));
  unsigned long long* keys = (unsigned long long*)take((size_t)B_ * NPTS * 8);
  int*   idxb   = (int*)take((size_t)B_ * NPTS * 4);
  hf*    featsS = (hf*)take((size_t)B_ * CIN * NPTS * sizeof(hf));
  float* xyzS   = (float*)take((size_t)B_ * 3 * NPTS * sizeof(float));
  float* rbuf   = (float*)take((size_t)B_ * COUT * NPTS * sizeof(float));
  hf*    ofeat  = (hf*)take((size_t)B_ * OUTC * NPTS * sizeof(hf));
  float* stats  = (float*)take((size_t)COUT * 2 * sizeof(float));

  cvt_f32_f16<<<(COUT * KDIM + 255) / 256, 256, 0, stream>>>(convw, wf16, COUT * KDIM);
  cvt_f32_f16<<<(OUTC * OUTC + 255) / 256, 256, 0, stream>>>(pww, pwf16, OUTC * OUTC);

  const unsigned bnBlocks  = (unsigned)((B_ * NPTS + 255) / 256);
  const unsigned bcnBlocks = (unsigned)(((size_t)B_ * COUT * NPTS + 255) / 256);

  for (int t = 0; t < TROT; ++t) {
    const float* rt = rots + t * 9;
    minmax_kernel<<<B_, 256, 0, stream>>>(coords, rt, mm);
    keys_kernel<<<bnBlocks, 256, 0, stream>>>(coords, rt, mm, dist, keys);
    for (int k = 2; k <= NPTS; k <<= 1)
      for (int j = k >> 1; j > 0; j >>= 1)
        bitonic_step<<<bnBlocks, 256, 0, stream>>>(keys, k, j);
    idx_kernel<<<bnBlocks, 256, 0, stream>>>(keys, idxb);
    gather_feats<<<bcnBlocks, 256, 0, stream>>>(x, idxb, featsS);
    gather_xyz<<<(B_ * 3 * NPTS + 255) / 256, 256, 0, stream>>>(coords, idxb, xyzS);
    conv_wmma<<<dim3(NPTS / NT, B_), 256, 0, stream>>>(featsS, xyzS, wf16, rbuf);
    bn_stats<<<COUT, 256, 0, stream>>>(rbuf, stats);
    bn_scatter<<<bcnBlocks, 256, 0, stream>>>(rbuf, stats, gamma, beta, idxb, ofeat, t);
  }
  pw_wmma<<<dim3(NPTS / 64, B_), 256, 0, stream>>>(ofeat, pwf16, pwb, out);
}
